// SelfFormer_block_10282151707112
// MI455X (gfx1250) — matrix operand order repaired and checked by reading, not compiled, not run on hardware
//
#include <hip/hip_runtime.h>
#include <math.h>

// ---------------- problem constants ----------------
#define Bn 2
#define Cc 64
#define HW 16384          // 128*128
#define Wd 128
#define Hh 128
#define NPIX (Bn*HW)      // 32768
#define NTOK 1536         // 48*32 tokens per grid cell
#define HPAD 192          // 128 + 64 top-pad rows

typedef __attribute__((ext_vector_type(16))) _Float16 v16h;
typedef __attribute__((ext_vector_type(8)))  float    v8f;

__device__ inline void cvt4(v16h& dst, int base, float4 v) {
  dst[base + 0] = (_Float16)v.x; dst[base + 1] = (_Float16)v.y;
  dst[base + 2] = (_Float16)v.z; dst[base + 3] = (_Float16)v.w;
}

// =====================================================================
// WMMA GEMM (NHWC):  Y[pix, m] = sum_k W[m,k] * X[pix, k] + bias[m]
// K compile-time (multiple of 32).  One wave computes a 16(M) x 64(N) tile:
// A loaded once per K-step (4x b128), 4 B tiles (4x b128 each), 8 static
// v_wmma_f32_16x16x32_f16, D stored as 2x b128 per tile per lane.
// Fragment layouts per ISA 7.12.2:
//   A 16x32 f16: lanes0-15 row m=l K{0..7,16..23}; lanes16-31 K{8..15,24..31}
//   B 32x16 f16: lane l col n=l, K kb..kb+7, kb+16..kb+23 (kb = half*8), same as A
//   C/D: VGPR r -> row m0 + r + half*8, col = lane&15
// =====================================================================
template<int K, int NT>
__global__ __launch_bounds__(32) void gemm_wmma_k(
    const float* __restrict__ Wm, const float* __restrict__ X,
    const float* __restrict__ bias, float* __restrict__ Y,
    int M, int relu)
{
  const int lane = threadIdx.x;
  const int l    = lane & 15;
  const int half = lane >> 4;
  const int n0   = blockIdx.x * (16 * NT);
  const int m0   = blockIdx.y * 16;
  const int mrow = m0 + l;

  // speculative prefetch of the next wave-tile's activations
  __builtin_prefetch(X + (size_t)(n0 + 16 * NT) * K, 0, 1);

  v8f acc[NT];
#pragma unroll
  for (int t = 0; t < NT; ++t) acc[t] = (v8f){};

  constexpr int KS = K / 32;
#pragma unroll
  for (int ks = 0; ks < KS; ++ks) {
    const int k0 = ks * 32;
    const int ka = k0 + half * 8;
    const int kb = k0 + half * 8;

    // A fragment: rows mrow, K = ka..ka+7 and ka+16..ka+23 (contiguous f32)
    const float4* Ap = (const float4*)(Wm + (size_t)mrow * K + ka);
    v16h a;
    cvt4(a, 0,  Ap[0]); cvt4(a, 4,  Ap[1]);
    cvt4(a, 8,  Ap[4]); cvt4(a, 12, Ap[5]);

#pragma unroll
    for (int t = 0; t < NT; ++t) {
      const int pix = n0 + t * 16 + l;
      const float4* Bp = (const float4*)(X + (size_t)pix * K + kb);
      v16h b;
      cvt4(b, 0, Bp[0]); cvt4(b, 4, Bp[1]); cvt4(b, 8, Bp[4]); cvt4(b, 12, Bp[5]);
      acc[t] = __builtin_amdgcn_wmma_f32_16x16x32_f16(
          false, a, false, b, (short)0, acc[t], false, false);
    }
  }

  const int mb = m0 + half * 8;
#pragma unroll
  for (int t = 0; t < NT; ++t) {
    const int pix = n0 + t * 16 + l;
    float4 o0, o1;
    o0.x = acc[t][0] + bias[mb + 0]; o0.y = acc[t][1] + bias[mb + 1];
    o0.z = acc[t][2] + bias[mb + 2]; o0.w = acc[t][3] + bias[mb + 3];
    o1.x = acc[t][4] + bias[mb + 4]; o1.y = acc[t][5] + bias[mb + 5];
    o1.z = acc[t][6] + bias[mb + 6]; o1.w = acc[t][7] + bias[mb + 7];
    if (relu) {
      o0.x = fmaxf(o0.x, 0.f); o0.y = fmaxf(o0.y, 0.f);
      o0.z = fmaxf(o0.z, 0.f); o0.w = fmaxf(o0.w, 0.f);
      o1.x = fmaxf(o1.x, 0.f); o1.y = fmaxf(o1.y, 0.f);
      o1.z = fmaxf(o1.z, 0.f); o1.w = fmaxf(o1.w, 0.f);
    }
    float4* Yp = (float4*)(Y + (size_t)pix * M + mb);
    Yp[0] = o0; Yp[1] = o1;
  }
}

// ---------------- small scalar GEMM (tiny M or K), NHWC ----------------
__global__ void gemm_small_k(const float* __restrict__ Wm, const float* __restrict__ X,
                             const float* __restrict__ bias, float* __restrict__ Y,
                             int M, int K)
{
  int idx = blockIdx.x * 256 + threadIdx.x;
  if (idx >= M * NPIX) return;
  int m = idx % M;
  int p = idx / M;
  const float* Xb = X + (size_t)p * K;
  float s = bias[m];
  for (int k = 0; k < K; ++k) s += Wm[m * K + k] * Xb[k];
  Y[(size_t)p * M + m] = s;
}

// ---------------- per-pixel LayerNorm over 64 channels (out: NHWC) -----------
__global__ void ln_k(const float* __restrict__ x, const float* __restrict__ w,
                     const float* __restrict__ bias, float* __restrict__ y,
                     int in_nchw)
{
  int p = blockIdx.x * 256 + threadIdx.x;
  if (p >= NPIX) return;
  int b = p >> 14, hw = p & (HW - 1);
  float s = 0.f, ss = 0.f;
  for (int c = 0; c < Cc; ++c) {
    float v = in_nchw ? x[((size_t)b * Cc + c) * HW + hw] : x[(size_t)p * Cc + c];
    s += v; ss += v * v;
  }
  float mu  = s * (1.f / Cc);
  float var = ss * (1.f / Cc) - mu * mu;
  float inv = rsqrtf(var + 1e-6f);
  float* yp = y + (size_t)p * Cc;
  for (int c = 0; c < Cc; ++c) {
    float v = in_nchw ? x[((size_t)b * Cc + c) * HW + hw] : x[(size_t)p * Cc + c];
    yp[c] = w[c] * ((v - mu) * inv) + bias[c];
  }
}

// ---------------- shifted depthwise 3x3 (DIL=1) + SimpleGate, NHWC -----------
// taps: rows h-2..h, cols w-1..w+1 (zeros outside); out[c]=dw(c)*dw(c+64)
__device__ inline float dw3x3(const float* __restrict__ ip, int h, int w,
                              const float* __restrict__ wt)
{
  float a = 0.f;
#pragma unroll
  for (int dh = 0; dh < 3; ++dh) {
    int hh = h - 2 + dh;
    if (hh < 0 || hh >= Hh) continue;
#pragma unroll
    for (int dwc = 0; dwc < 3; ++dwc) {
      int ww = w - 1 + dwc;
      if (ww < 0 || ww >= Wd) continue;
      a += wt[dh * 3 + dwc] * ip[(size_t)(hh * Wd + ww) * 128];
    }
  }
  return a;
}

__global__ void dwgate_k(const float* __restrict__ in, const float* __restrict__ w9,
                         const float* __restrict__ bias, float* __restrict__ out)
{
  int idx = blockIdx.x * 256 + threadIdx.x;          // B*HW*64
  if (idx >= Bn * HW * Cc) return;
  int c = idx & 63, w = (idx >> 6) & 127, h = (idx >> 13) & 127, b = idx >> 20;
  const float* slab = in + (size_t)b * HW * 128;
  float r0 = dw3x3(slab + c,      h, w, w9 + c * 9)        + bias[c];
  float r1 = dw3x3(slab + c + 64, h, w, w9 + (c + 64) * 9) + bias[c + 64];
  out[idx] = r0 * r1;
}

// ---------------- SCA: pool (NHWC, coalesced), 64x64 matmul, scale -----------
__global__ void pool_k(const float* __restrict__ in, float* __restrict__ pooled)
{
  __shared__ float sm[256];
  int b = blockIdx.x >> 4;
  int chunk = blockIdx.x & 15;                       // 1024 pixels per chunk
  int c = threadIdx.x & 63;
  int pl = threadIdx.x >> 6;                         // 0..3
  float s = 0.f;
  for (int q = 0; q < 1024; q += 4) {
    int pix = chunk * 1024 + q + pl;
    s += in[((size_t)b * HW + pix) * 64 + c];
  }
  sm[threadIdx.x] = s; __syncthreads();
  if (threadIdx.x < 64) {
    float tot = sm[threadIdx.x] + sm[threadIdx.x + 64] +
                sm[threadIdx.x + 128] + sm[threadIdx.x + 192];
    atomicAdd(&pooled[b * 64 + threadIdx.x], tot * (1.f / HW));
  }
}

__global__ void sca_k(const float* __restrict__ pooled, const float* __restrict__ scaw,
                      const float* __restrict__ scab, float* __restrict__ svec)
{
  int t = threadIdx.x; if (t >= 128) return;
  int b = t >> 6, co = t & 63;
  float s = scab[co];
  for (int ci = 0; ci < 64; ++ci) s += scaw[co * 64 + ci] * pooled[b * 64 + ci];
  svec[t] = s;
}

__global__ void scale_k(float* __restrict__ x, const float* __restrict__ svec)
{
  int idx = blockIdx.x * 256 + threadIdx.x;
  if (idx >= Bn * HW * Cc) return;
  int c = idx & 63, b = idx >> 20;
  x[idx] *= svec[b * 64 + c];
}

// ---------------- SimpleGate on a 128-ch NHWC tensor ----------------
__global__ void gate_k(const float* __restrict__ in, float* __restrict__ out)
{
  int idx = blockIdx.x * 256 + threadIdx.x;
  if (idx >= Bn * HW * Cc) return;
  int c = idx & 63, p = idx >> 6;
  out[idx] = in[(size_t)p * 128 + c] * in[(size_t)p * 128 + 64 + c];
}

// ------- residual: out = A + B*scale[c]; A may be NCHW, out may be NCHW ------
__global__ void resid_k(const float* __restrict__ A, const float* __restrict__ Bt,
                        const float* __restrict__ scale, float* __restrict__ out,
                        int a_nchw, int out_nchw)
{
  int idx = blockIdx.x * 256 + threadIdx.x;          // NHWC index space
  if (idx >= Bn * HW * Cc) return;
  int c = idx & 63, p = idx >> 6, b = p >> 14, hw = p & (HW - 1);
  float av = a_nchw ? A[((size_t)b * Cc + c) * HW + hw] : A[idx];
  float sc = scale ? scale[c] : 1.f;
  float v = av + Bt[idx] * sc;
  if (out_nchw) out[((size_t)b * Cc + c) * HW + hw] = v;
  else          out[idx] = v;
}

// ---------------- attention: per-batch sum/sumsq of 4-ch NHWC tensor ---------
__global__ void stats_k(const float* __restrict__ xa, float* __restrict__ stats)
{
  __shared__ float s1[256], s2[256];
  int idx = blockIdx.x * 256 + threadIdx.x;          // B*HW*4 = 131072
  float v = xa[idx];
  s1[threadIdx.x] = v; s2[threadIdx.x] = v * v;
  __syncthreads();
  for (int off = 128; off; off >>= 1) {
    if (threadIdx.x < off) {
      s1[threadIdx.x] += s1[threadIdx.x + off];
      s2[threadIdx.x] += s2[threadIdx.x + off];
    }
    __syncthreads();
  }
  if (threadIdx.x == 0) {
    int b = idx >> 16;                               // 4*HW = 2^16 per batch
    atomicAdd(&stats[b * 2],     s1[0]);
    atomicAdd(&stats[b * 2 + 1], s2[0]);
  }
}

// ------- build top-padded, globally-normalized xx (B,192,128,4 NHWC) ---------
__global__ void xx_k(const float* __restrict__ xa, const float* __restrict__ stats,
                     float* __restrict__ xx)
{
  int idx = blockIdx.x * 256 + threadIdx.x;          // B*192*128*4 = 196608
  if (idx >= Bn * HPAD * Wd * 4) return;
  int c = idx & 3;
  int rest = idx >> 2;
  int w = rest & 127; rest >>= 7;
  int h = rest % HPAD;
  int b = rest / HPAD;
  const float Np = 4.f * HPAD * Wd;                  // 98304 incl. zero pad
  float mean = stats[b * 2] / Np;
  float var  = stats[b * 2 + 1] / Np - mean * mean;
  float inv  = rsqrtf(var + 1e-5f);
  float v = (h < 64) ? 0.f
          : xa[((size_t)b * HW + (h - 64) * Wd + w) * 4 + c];
  xx[idx] = (v - mean) * inv;
}

// ---------------- grid attention core: one block per (b,i,j) grid cell -------
__global__ __launch_bounds__(256) void att_core_k(
    const float* __restrict__ xx, const float* __restrict__ wqk,
    float* __restrict__ att)
{
  __shared__ float tokS[NTOK * 4];
  __shared__ float qkS[NTOK * 4];
  int gid = blockIdx.x;
  int b = gid >> 4, i = (gid >> 2) & 3, j = gid & 3;

  float wq[16];
#pragma unroll
  for (int t = 0; t < 16; ++t) wq[t] = wqk[t];

  for (int n = threadIdx.x; n < NTOK; n += 256) {
    int r = n >> 5, s = n & 31;
    int h = r * 4 + i, w = s * 4 + j;
    float4 tv = *(const float4*)(xx + ((size_t)(b * HPAD + h) * Wd + w) * 4);
    tokS[n * 4 + 0] = tv.x; tokS[n * 4 + 1] = tv.y;
    tokS[n * 4 + 2] = tv.z; tokS[n * 4 + 3] = tv.w;
    float q[4];
#pragma unroll
    for (int k = 0; k < 4; ++k)
      q[k] = tv.x * wq[k] + tv.y * wq[4 + k] + tv.z * wq[8 + k] + tv.w * wq[12 + k];
    float nrm = sqrtf(q[0]*q[0] + q[1]*q[1] + q[2]*q[2] + q[3]*q[3]) + 1e-8f;
    float inn = 1.f / nrm;
#pragma unroll
    for (int k = 0; k < 4; ++k) qkS[n * 4 + k] = q[k] * inn;
  }
  __syncthreads();

  const float sc = 0.17677669529663687f;             // 1/sqrt(32)
  int wave = threadIdx.x >> 5, lane = threadIdx.x & 31;
  for (int n = wave; n < 1024; n += 8) {             // only rows with h < 128
    float q0 = qkS[n*4], q1 = qkS[n*4+1], q2 = qkS[n*4+2], q3 = qkS[n*4+3];
    float den = 0.f, a0 = 0.f, a1 = 0.f, a2 = 0.f, a3 = 0.f;
    for (int m = lane; m < NTOK; m += 32) {
      float d = q0*qkS[m*4] + q1*qkS[m*4+1] + q2*qkS[m*4+2] + q3*qkS[m*4+3];
      float e = __expf((d + 1.f) * sc);              // bounded logits: no max
      den += e;
      a0 += e * tokS[m*4];   a1 += e * tokS[m*4+1];
      a2 += e * tokS[m*4+2]; a3 += e * tokS[m*4+3];
    }
    for (int off = 16; off; off >>= 1) {
      den += __shfl_xor(den, off, 32);
      a0 += __shfl_xor(a0, off, 32); a1 += __shfl_xor(a1, off, 32);
      a2 += __shfl_xor(a2, off, 32); a3 += __shfl_xor(a3, off, 32);
    }
    if (lane == 0) {
      int r = n >> 5, s = n & 31;
      int h = r * 4 + i, w = s * 4 + j;              // h <= 127 guaranteed
      float inv = 1.f / den;
      float4 o; o.x = a0 * inv; o.y = a1 * inv; o.z = a2 * inv; o.w = a3 * inv;
      *(float4*)(att + ((size_t)b * HW + h * Wd + w) * 4) = o;
    }
  }
}

// -------- FFN mid conv: dense 16->16 3x3, dil=2, shift s=3, relu (NHWC) ------
// taps: rows {h-5,h-3,h-1}, cols {w-2,w,w+2}
__global__ void ffconv_k(const float* __restrict__ in, const float* __restrict__ wt,
                         const float* __restrict__ bias, float* __restrict__ out)
{
  int idx = blockIdx.x * 256 + threadIdx.x;          // B*HW*16 = 524288
  if (idx >= Bn * HW * 16) return;
  int o = idx & 15, w = (idx >> 4) & 127, h = (idx >> 11) & 127, b = idx >> 18;
  const float* slab = in + (size_t)b * HW * 16;
  float acc = bias[o];
#pragma unroll
  for (int dh = 0; dh < 3; ++dh) {
    int hh = h - 5 + 2 * dh;
    if (hh < 0 || hh >= Hh) continue;
#pragma unroll
    for (int dwc = 0; dwc < 3; ++dwc) {
      int ww = w - 2 + 2 * dwc;
      if (ww < 0 || ww >= Wd) continue;
      const float* px = slab + (size_t)(hh * Wd + ww) * 16;
      for (int ic = 0; ic < 16; ++ic)
        acc += wt[(o * 16 + ic) * 9 + dh * 3 + dwc] * px[ic];
    }
  }
  out[idx] = fmaxf(acc, 0.f);
}

// =====================================================================
extern "C" void kernel_launch(void* const* d_in, const int* in_sizes, int n_in,
                              void* d_out, int out_size, void* d_ws, size_t ws_size,
                              hipStream_t stream)
{
  (void)in_sizes; (void)n_in; (void)out_size; (void)ws_size;
  const float* x    = (const float*)d_in[0];
  const float* n1w  = (const float*)d_in[1];
  const float* n1b  = (const float*)d_in[2];
  const float* nc1w = (const float*)d_in[3];
  const float* nc1b = (const float*)d_in[4];
  const float* nc2w = (const float*)d_in[5];
  const float* nc2b = (const float*)d_in[6];
  const float* scaw = (const float*)d_in[7];
  const float* scab = (const float*)d_in[8];
  const float* nc3w = (const float*)d_in[9];
  const float* nc3b = (const float*)d_in[10];
  const float* n2w  = (const float*)d_in[11];
  const float* n2b  = (const float*)d_in[12];
  const float* nc4w = (const float*)d_in[13];
  const float* nc4b = (const float*)d_in[14];
  const float* nc5w = (const float*)d_in[15];
  const float* nc5b = (const float*)d_in[16];
  const float* beta = (const float*)d_in[17];
  const float* gamma= (const float*)d_in[18];
  const float* wqk  = (const float*)d_in[19];
  const float* a1aw = (const float*)d_in[20];
  const float* a1ab = (const float*)d_in[21];
  const float* a1bw = (const float*)d_in[22];
  const float* a1bb = (const float*)d_in[23];
  const float* a2w  = (const float*)d_in[24];
  const float* a2b  = (const float*)d_in[25];
  const float* acw  = (const float*)d_in[26];
  const float* acb  = (const float*)d_in[27];
  const float* g1w  = (const float*)d_in[28];
  const float* g1b  = (const float*)d_in[29];
  const float* g2w  = (const float*)d_in[30];
  const float* g2b  = (const float*)d_in[31];
  const float* g3w  = (const float*)d_in[32];
  const float* g3b  = (const float*)d_in[33];

  // workspace layout (floats); all activations NHWC
  float* ws   = (float*)d_ws;
  float* t64a = ws;
  float* t64b = t64a + (size_t)Cc * NPIX;
  float* t64c = t64b + (size_t)Cc * NPIX;       // y  (first residual)
  float* t64d = t64c + (size_t)Cc * NPIX;       // y1 (naf output)
  float* t128 = t64d + (size_t)Cc * NPIX;       // 128-ch scratch
  float* t16a = t128 + (size_t)2 * Cc * NPIX;
  float* t16b = t16a + (size_t)16 * NPIX;
  float* t4a  = t16b + (size_t)16 * NPIX;
  float* t4b  = t4a + (size_t)4 * NPIX;
  float* xx   = t4b + (size_t)4 * NPIX;         // B*192*128*4
  float* pooled = xx + (size_t)Bn * HPAD * Wd * 4;
  float* svec = pooled + 128;
  float* stats = svec + 128;

  const int EW = (Bn * HW * Cc) / 256;          // 8192 blocks
  const dim3 GG(NPIX / 64, 0, 0);               // gemm grid.x = 512

  hipMemsetAsync(pooled, 0, 260 * sizeof(float), stream);  // pooled+svec+stats

  // ---------------- NAF block ----------------
  ln_k<<<NPIX / 256, 256, 0, stream>>>(x, n1w, n1b, t64a, 1);
  gemm_wmma_k<64, 4><<<dim3(NPIX/64, 8), 32, 0, stream>>>(nc1w, t64a, nc1b, t128, 128, 0);
  dwgate_k<<<EW, 256, 0, stream>>>(t128, nc2w, nc2b, t64b);
  pool_k<<<32, 256, 0, stream>>>(t64b, pooled);
  sca_k<<<1, 128, 0, stream>>>(pooled, scaw, scab, svec);
  scale_k<<<EW, 256, 0, stream>>>(t64b, svec);
  gemm_wmma_k<64, 4><<<dim3(NPIX/64, 4), 32, 0, stream>>>(nc3w, t64b, nc3b, t64a, 64, 0);
  resid_k<<<EW, 256, 0, stream>>>(x, t64a, beta, t64c, 1, 0);           // y
  ln_k<<<NPIX / 256, 256, 0, stream>>>(t64c, n2w, n2b, t64a, 0);
  gemm_wmma_k<64, 4><<<dim3(NPIX/64, 8), 32, 0, stream>>>(nc4w, t64a, nc4b, t128, 128, 0);
  gate_k<<<EW, 256, 0, stream>>>(t128, t64b);
  gemm_wmma_k<64, 4><<<dim3(NPIX/64, 4), 32, 0, stream>>>(nc5w, t64b, nc5b, t64a, 64, 0);
  resid_k<<<EW, 256, 0, stream>>>(t64c, t64a, gamma, t64d, 0, 0);       // y1

  // ---------------- grid attention ----------------
  gemm_small_k<<<(4 * NPIX) / 256, 256, 0, stream>>>(a1aw, t64d, a1ab, t4a, 4, 64);
  gemm_small_k<<<(4 * NPIX) / 256, 256, 0, stream>>>(a1bw, t4a, a1bb, t4b, 4, 4);
  stats_k<<<(Bn * 4 * HW) / 256, 256, 0, stream>>>(t4b, stats);
  xx_k<<<(Bn * HPAD * Wd * 4) / 256, 256, 0, stream>>>(t4b, stats, xx);
  att_core_k<<<32, 256, 0, stream>>>(xx, wqk, t4a);                     // att NHWC
  gemm_small_k<<<(Cc * NPIX) / 256, 256, 0, stream>>>(a2w, t4a, a2b, t64a, 64, 4);
  gemm_wmma_k<64, 4><<<dim3(NPIX/64, 4), 32, 0, stream>>>(acw, t64a, acb, t64b, 64, 0);
  resid_k<<<EW, 256, 0, stream>>>(t64d, t64b, nullptr, t64a, 0, 0);     // z

  // FFN
  gemm_wmma_k<64, 4><<<dim3(NPIX/64, 1), 32, 0, stream>>>(g1w, t64a, g1b, t16a, 16, 1);
  ffconv_k<<<(Bn * HW * 16) / 256, 256, 0, stream>>>(t16a, g2w, g2b, t16b);
  gemm_small_k<<<(Cc * NPIX) / 256, 256, 0, stream>>>(g3w, t16b, g3b, t64b, 64, 16);
  resid_k<<<EW, 256, 0, stream>>>(t64d, t64b, nullptr, (float*)d_out, 0, 1);
  (void)GG;
}
